// Seq2SeqTranslator_9328668967702
// MI455X (gfx1250) — matrix operand order repaired and checked by reading, not compiled, not run on hardware
//
#include <hip/hip_runtime.h>
#include <math.h>

typedef __bf16 bf16;
typedef bf16  v16bf __attribute__((ext_vector_type(16)));
typedef bf16  v8bf  __attribute__((ext_vector_type(8)));
typedef float v8f   __attribute__((ext_vector_type(8)));

#define BB 32
#define SS 64
#define TT 64
#define DD 256
#define HH 512
#define VV 32000
#define GG (3*HH)

// ---------------------------------------------------------------------------
// Utility kernels
// ---------------------------------------------------------------------------
__global__ void k_cvt_f32_bf16(const float* __restrict__ src, bf16* __restrict__ dst, int n) {
  int i = blockIdx.x * blockDim.x + threadIdx.x;
  if (i < n) dst[i] = (bf16)src[i];
}

__global__ void k_zero_f32(float* __restrict__ p, int n) {
  int i = blockIdx.x * blockDim.x + threadIdx.x;
  if (i < n) p[i] = 0.0f;
}

__global__ void k_zero_bf16(bf16* __restrict__ p, int n) {
  int i = blockIdx.x * blockDim.x + threadIdx.x;
  if (i < n) p[i] = (bf16)0.0f;
}

// Encoder embedding gather: out[s,b,d] = bf16(E_enc[input_seq[b,s], d])
__global__ void k_gather_enc(const int* __restrict__ seq, const float* __restrict__ E,
                             bf16* __restrict__ out) {
  int i = blockIdx.x * blockDim.x + threadIdx.x;
  if (i >= SS * BB * DD) return;
  int d = i % DD;
  int sb = i / DD;
  int b = sb % BB;
  int s = sb / BB;
  int tok = seq[b * SS + s];
  out[i] = (bf16)E[(size_t)tok * DD + d];
}

// Decoder embedding gather with teacher-forcing shift:
// token(t=0)=0, token(t>0)=target[b,t-1]; out[t,b,d]
__global__ void k_gather_dec(const int* __restrict__ tgt, const float* __restrict__ E,
                             bf16* __restrict__ out) {
  int i = blockIdx.x * blockDim.x + threadIdx.x;
  if (i >= TT * BB * DD) return;
  int d = i % DD;
  int tb = i / DD;
  int b = tb % BB;
  int t = tb / BB;
  int tok = (t == 0) ? 0 : tgt[b * TT + t - 1];
  out[i] = (bf16)E[(size_t)tok * DD + d];
}

// ---------------------------------------------------------------------------
// WMMA GEMM:  C[M,N](f32) = A[M,K](bf16,row-major,lda) @ W[N,K]^T (bf16,row-major,ldw) + bias
// One wave computes one 32x16 output tile (two 16x16 M-tiles sharing a single
// B fragment -> W stream read once instead of twice). K consumed 32/iteration
// via v_wmma_f32_16x16x32_bf16 (two per K-step). Requires M % 32 == 0.
//
// A fragment (16x32 bf16, ISA 7.12.2): lane = M row (mod 16); lanes<16 hold
// K[0..7]+K[16..23], lanes>=16 hold K[8..15]+K[24..31] -> two 16B loads.
// B fragment (32x16 bf16): lane = N col (mod 16); same K split as A
// (lanes<16 K[0..7]+K[16..23], lanes>=16 K[8..15]+K[24..31]) -> two 16B loads.
// C/D: lane<16 -> N=lane, M=r; lane>=16 -> N=lane-16, M=r+8.
// ---------------------------------------------------------------------------
__global__ void k_gemm_bf16_wmma(const bf16* __restrict__ A, int lda,
                                 const bf16* __restrict__ W, int ldw,
                                 const float* __restrict__ bias,
                                 float* __restrict__ C, long long ldc,
                                 int M, int N, int K) {
  int wave = (int)((blockIdx.x * blockDim.x + threadIdx.x) >> 5);
  int lane = threadIdx.x & 31;
  int tilesN = N >> 4;
  int total = (M >> 5) * tilesN;
  if (wave >= total) return;

  int tm = wave / tilesN;
  int tn = wave - tm * tilesN;
  int m0 = tm << 5;               // 32 rows per wave
  int n0 = tn << 4;
  int half = lane >> 4;           // 0 or 1
  int l16  = lane & 15;

  const bf16* arow0 = A + (size_t)(m0 + l16) * (size_t)lda + half * 8;
  const bf16* arow1 = arow0 + (size_t)16 * (size_t)lda;
  const bf16* wrow  = W + (size_t)(n0 + l16) * (size_t)ldw + half * 8;

  v8f acc0 = {};
  v8f acc1 = {};
  for (int k = 0; k < K; k += 32) {
    __builtin_prefetch((const void*)(wrow + k + 128), 0, 0);
    v8bf blo = *(const v8bf*)(wrow + k);
    v8bf bhi = *(const v8bf*)(wrow + k + 16);
    v16bf b = __builtin_shufflevector(blo, bhi,
                                      0, 1, 2, 3, 4, 5, 6, 7,
                                      8, 9, 10, 11, 12, 13, 14, 15);

    v8bf a0lo = *(const v8bf*)(arow0 + k);
    v8bf a0hi = *(const v8bf*)(arow0 + k + 16);
    v16bf a0 = __builtin_shufflevector(a0lo, a0hi,
                                       0, 1, 2, 3, 4, 5, 6, 7,
                                       8, 9, 10, 11, 12, 13, 14, 15);
    acc0 = __builtin_amdgcn_wmma_f32_16x16x32_bf16(false, a0, false, b,
                                                   (short)0, acc0, false, false);

    v8bf a1lo = *(const v8bf*)(arow1 + k);
    v8bf a1hi = *(const v8bf*)(arow1 + k + 16);
    v16bf a1 = __builtin_shufflevector(a1lo, a1hi,
                                       0, 1, 2, 3, 4, 5, 6, 7,
                                       8, 9, 10, 11, 12, 13, 14, 15);
    acc1 = __builtin_amdgcn_wmma_f32_16x16x32_bf16(false, a1, false, b,
                                                   (short)0, acc1, false, false);
  }

  float bv = bias ? bias[n0 + l16] : 0.0f;
  float* crow0 = C + (size_t)(m0 + half * 8) * (size_t)ldc + (n0 + l16);
  float* crow1 = crow0 + (size_t)16 * (size_t)ldc;
#pragma unroll
  for (int r = 0; r < 8; ++r) {
    crow0[(size_t)r * (size_t)ldc] = acc0[r] + bv;
    crow1[(size_t)r * (size_t)ldc] = acc1[r] + bv;
  }
}

// ---------------------------------------------------------------------------
// Fused GRU gate kernel (PyTorch gate order r,z,n):
//   r=sig(i_r+h_r); z=sig(i_z+h_z); n=tanh(i_n+r*h_n); h'=(1-z)*n+z*h
// Updates h (f32, in-place) + bf16 mirror, optionally streams h' into a
// sequence buffer (f32 and/or bf16) with per-batch stride.
// ---------------------------------------------------------------------------
__device__ __forceinline__ float sigf(float x) { return 1.0f / (1.0f + expf(-x)); }

__global__ void k_gru_gates(const float* __restrict__ gi, const float* __restrict__ gh,
                            float* __restrict__ h, bf16* __restrict__ hb,
                            float* __restrict__ seqf, long long seqf_strideB,
                            bf16* __restrict__ seqb, long long seqb_strideB) {
  int i = blockIdx.x * blockDim.x + threadIdx.x;
  if (i >= BB * HH) return;
  int b = i / HH;
  int j = i - b * HH;
  const float* gib = gi + (size_t)b * GG;
  const float* ghb = gh + (size_t)b * GG;
  float r = sigf(gib[j] + ghb[j]);
  float z = sigf(gib[HH + j] + ghb[HH + j]);
  float n = tanhf(gib[2 * HH + j] + r * ghb[2 * HH + j]);
  float hn = (1.0f - z) * n + z * h[i];
  h[i] = hn;
  hb[i] = (bf16)hn;
  if (seqf) seqf[(size_t)b * (size_t)seqf_strideB + j] = hn;
  if (seqb) seqb[(size_t)b * (size_t)seqb_strideB + j] = (bf16)hn;
}

// ---------------------------------------------------------------------------
// Attention: scores[b,s] = sum_j v[j] * tanh(hproj[b,j] + encproj[b*S+s, j])
// One wave per (b,s); wave reduction via shfl_xor (wave32).
// ---------------------------------------------------------------------------
__global__ void k_attn_scores(const float* __restrict__ hproj,
                              const float* __restrict__ encproj,
                              const float* __restrict__ v,
                              float* __restrict__ scores) {
  int wave = (int)((blockIdx.x * blockDim.x + threadIdx.x) >> 5);
  int lane = threadIdx.x & 31;
  if (wave >= BB * SS) return;
  int b = wave / SS;
  const float* ep = encproj + (size_t)wave * HH;
  const float* hp = hproj + (size_t)b * HH;
  float sum = 0.0f;
  for (int j = lane; j < HH; j += 32)
    sum += v[j] * tanhf(hp[j] + ep[j]);
#pragma unroll
  for (int off = 16; off > 0; off >>= 1)
    sum += __shfl_xor(sum, off, 32);
  if (lane == 0) scores[wave] = sum;
}

// Row softmax over S=64 per batch row (B=32 rows -> one thread each)
__global__ void k_softmax_rows(const float* __restrict__ scores, float* __restrict__ a) {
  int b = blockIdx.x * blockDim.x + threadIdx.x;
  if (b >= BB) return;
  const float* in = scores + (size_t)b * SS;
  float* out = a + (size_t)b * SS;
  float mx = -3.0e38f;
  for (int s = 0; s < SS; ++s) mx = fmaxf(mx, in[s]);
  float sum = 0.0f;
  for (int s = 0; s < SS; ++s) { float e = expf(in[s] - mx); out[s] = e; sum += e; }
  float inv = 1.0f / sum;
  for (int s = 0; s < SS; ++s) out[s] *= inv;
}

// ctx[b,j] = sum_s a[b,s] * enc_out[b,s,j]
__global__ void k_attn_ctx(const float* __restrict__ a, const float* __restrict__ encout,
                           float* __restrict__ ctx) {
  int i = blockIdx.x * blockDim.x + threadIdx.x;
  if (i >= BB * HH) return;
  int b = i / HH;
  int j = i - b * HH;
  float s = 0.0f;
  for (int t = 0; t < SS; ++t)
    s += a[(size_t)b * SS + t] * encout[((size_t)b * SS + t) * HH + j];
  ctx[i] = s;
}

// rnn_in[b, :] = [dec_emb_t[b, 0:D] (bf16), bf16(ctx[b, 0:H])]
__global__ void k_build_rnn_in(const bf16* __restrict__ demb_t, const float* __restrict__ ctx,
                               bf16* __restrict__ out) {
  int i = blockIdx.x * blockDim.x + threadIdx.x;
  const int W = DD + HH;
  if (i >= BB * W) return;
  int b = i / W;
  int j = i - b * W;
  out[i] = (j < DD) ? demb_t[(size_t)b * DD + j] : (bf16)ctx[(size_t)b * HH + (j - DD)];
}

// ---------------------------------------------------------------------------
// Host side
// ---------------------------------------------------------------------------
static inline char* arena_alloc(char*& p, size_t bytes) {
  char* r = p;
  p += (bytes + 255) & ~(size_t)255;
  return r;
}

static void launch_gemm(const bf16* A, int lda, const bf16* W, int ldw,
                        const float* bias, float* C, long long ldc,
                        int M, int N, int K, hipStream_t stream) {
  int waves = (M / 32) * (N / 16);   // 32x16 tile per wave
  int threads = waves * 32;
  int block = 256;
  int grid = (threads + block - 1) / block;
  k_gemm_bf16_wmma<<<grid, block, 0, stream>>>(A, lda, W, ldw, bias, C, ldc, M, N, K);
}

extern "C" void kernel_launch(void* const* d_in, const int* in_sizes, int n_in,
                              void* d_out, int out_size, void* d_ws, size_t ws_size,
                              hipStream_t stream) {
  (void)in_sizes; (void)n_in; (void)out_size; (void)ws_size;

  const int*   input_seq  = (const int*)  d_in[0];
  const int*   target_seq = (const int*)  d_in[1];
  const float* E_enc      = (const float*)d_in[2];
  const float* Wx_e0      = (const float*)d_in[3];
  const float* Wh_e0      = (const float*)d_in[4];
  const float* bx_e0      = (const float*)d_in[5];
  const float* bh_e0      = (const float*)d_in[6];
  const float* Wx_e1      = (const float*)d_in[7];
  const float* Wh_e1      = (const float*)d_in[8];
  const float* bx_e1      = (const float*)d_in[9];
  const float* bh_e1      = (const float*)d_in[10];
  const float* E_dec      = (const float*)d_in[11];
  const float* W_attn     = (const float*)d_in[12];
  const float* b_attn     = (const float*)d_in[13];
  const float* v_attn     = (const float*)d_in[14];
  const float* Wx_d0      = (const float*)d_in[15];
  const float* Wh_d0      = (const float*)d_in[16];
  const float* bx_d0      = (const float*)d_in[17];
  const float* bh_d0      = (const float*)d_in[18];
  const float* Wx_d1      = (const float*)d_in[19];
  const float* Wh_d1      = (const float*)d_in[20];
  const float* bx_d1      = (const float*)d_in[21];
  const float* bh_d1      = (const float*)d_in[22];
  const float* W_out      = (const float*)d_in[23];
  const float* b_out      = (const float*)d_in[24];
  float* out = (float*)d_out;

  // ---- workspace carve-up ----
  char* p = (char*)d_ws;
  bf16* wxe0b  = (bf16*)arena_alloc(p, sizeof(bf16) * (size_t)GG * DD);
  bf16* whe0b  = (bf16*)arena_alloc(p, sizeof(bf16) * (size_t)GG * HH);
  bf16* wxe1b  = (bf16*)arena_alloc(p, sizeof(bf16) * (size_t)GG * HH);
  bf16* whe1b  = (bf16*)arena_alloc(p, sizeof(bf16) * (size_t)GG * HH);
  bf16* wattnb = (bf16*)arena_alloc(p, sizeof(bf16) * (size_t)HH * 2 * HH);
  bf16* wxd0b  = (bf16*)arena_alloc(p, sizeof(bf16) * (size_t)GG * (DD + HH));
  bf16* whd0b  = (bf16*)arena_alloc(p, sizeof(bf16) * (size_t)GG * HH);
  bf16* wxd1b  = (bf16*)arena_alloc(p, sizeof(bf16) * (size_t)GG * HH);
  bf16* whd1b  = (bf16*)arena_alloc(p, sizeof(bf16) * (size_t)GG * HH);
  bf16* woutb  = (bf16*)arena_alloc(p, sizeof(bf16) * (size_t)VV * HH);

  bf16*  embb    = (bf16*) arena_alloc(p, sizeof(bf16)  * (size_t)SS * BB * DD);
  bf16*  decembb = (bf16*) arena_alloc(p, sizeof(bf16)  * (size_t)TT * BB * DD);
  float* gi0     = (float*)arena_alloc(p, sizeof(float) * (size_t)SS * BB * GG);
  float* gi1     = (float*)arena_alloc(p, sizeof(float) * (size_t)SS * BB * GG);
  bf16*  y0b     = (bf16*) arena_alloc(p, sizeof(bf16)  * (size_t)SS * BB * HH);
  float* encout  = (float*)arena_alloc(p, sizeof(float) * (size_t)BB * SS * HH);
  bf16*  encoutb = (bf16*) arena_alloc(p, sizeof(bf16)  * (size_t)BB * SS * HH);
  float* encproj = (float*)arena_alloc(p, sizeof(float) * (size_t)BB * SS * HH);
  float* h0      = (float*)arena_alloc(p, sizeof(float) * (size_t)BB * HH);
  float* h1      = (float*)arena_alloc(p, sizeof(float) * (size_t)BB * HH);
  bf16*  h0b     = (bf16*) arena_alloc(p, sizeof(bf16)  * (size_t)BB * HH);
  bf16*  h1b     = (bf16*) arena_alloc(p, sizeof(bf16)  * (size_t)BB * HH);
  float* gi_s    = (float*)arena_alloc(p, sizeof(float) * (size_t)BB * GG);
  float* gh_s    = (float*)arena_alloc(p, sizeof(float) * (size_t)BB * GG);
  float* hproj   = (float*)arena_alloc(p, sizeof(float) * (size_t)BB * HH);
  float* scores  = (float*)arena_alloc(p, sizeof(float) * (size_t)BB * SS);
  float* attw    = (float*)arena_alloc(p, sizeof(float) * (size_t)BB * SS);
  float* ctx     = (float*)arena_alloc(p, sizeof(float) * (size_t)BB * HH);
  bf16*  rnninb  = (bf16*) arena_alloc(p, sizeof(bf16)  * (size_t)BB * (DD + HH));

  const int blk = 256;
#define NB(n) (((n) + blk - 1) / blk)

  // ---- convert weights to bf16 (weights stay L2-resident afterwards) ----
  k_cvt_f32_bf16<<<NB(GG * DD), blk, 0, stream>>>(Wx_e0, wxe0b, GG * DD);
  k_cvt_f32_bf16<<<NB(GG * HH), blk, 0, stream>>>(Wh_e0, whe0b, GG * HH);
  k_cvt_f32_bf16<<<NB(GG * HH), blk, 0, stream>>>(Wx_e1, wxe1b, GG * HH);
  k_cvt_f32_bf16<<<NB(GG * HH), blk, 0, stream>>>(Wh_e1, whe1b, GG * HH);
  k_cvt_f32_bf16<<<NB(HH * 2 * HH), blk, 0, stream>>>(W_attn, wattnb, HH * 2 * HH);
  k_cvt_f32_bf16<<<NB(GG * (DD + HH)), blk, 0, stream>>>(Wx_d0, wxd0b, GG * (DD + HH));
  k_cvt_f32_bf16<<<NB(GG * HH), blk, 0, stream>>>(Wh_d0, whd0b, GG * HH);
  k_cvt_f32_bf16<<<NB(GG * HH), blk, 0, stream>>>(Wx_d1, wxd1b, GG * HH);
  k_cvt_f32_bf16<<<NB(GG * HH), blk, 0, stream>>>(Wh_d1, whd1b, GG * HH);
  k_cvt_f32_bf16<<<NB(VV * HH), blk, 0, stream>>>(W_out, woutb, VV * HH);

  // ---- embeddings ----
  k_gather_enc<<<NB(SS * BB * DD), blk, 0, stream>>>(input_seq, E_enc, embb);
  k_gather_dec<<<NB(TT * BB * DD), blk, 0, stream>>>(target_seq, E_dec, decembb);

  // ---- init hidden states ----
  k_zero_f32<<<NB(BB * HH), blk, 0, stream>>>(h0, BB * HH);
  k_zero_f32<<<NB(BB * HH), blk, 0, stream>>>(h1, BB * HH);
  k_zero_bf16<<<NB(BB * HH), blk, 0, stream>>>(h0b, BB * HH);
  k_zero_bf16<<<NB(BB * HH), blk, 0, stream>>>(h1b, BB * HH);

  // ---- encoder layer 0: batched input-gates GEMM, then sequential recurrence ----
  // gi0[s*B+b, :] = emb @ Wx_e0^T + bx_e0   (M=2048, N=1536, K=256)
  launch_gemm(embb, DD, wxe0b, DD, bx_e0, gi0, GG, SS * BB, GG, DD, stream);
  for (int s = 0; s < SS; ++s) {
    launch_gemm(h0b, HH, whe0b, HH, bh_e0, gh_s, GG, BB, GG, HH, stream);
    k_gru_gates<<<NB(BB * HH), blk, 0, stream>>>(
        gi0 + (size_t)s * BB * GG, gh_s, h0, h0b,
        nullptr, 0, y0b + (size_t)s * BB * HH, (long long)HH);
  }

  // ---- encoder layer 1 ----
  launch_gemm(y0b, HH, wxe1b, HH, bx_e1, gi1, GG, SS * BB, GG, HH, stream);
  for (int s = 0; s < SS; ++s) {
    launch_gemm(h1b, HH, whe1b, HH, bh_e1, gh_s, GG, BB, GG, HH, stream);
    // stream h into enc_out[b,s,:] (f32 + bf16), per-batch stride S*H
    k_gru_gates<<<NB(BB * HH), blk, 0, stream>>>(
        gi1 + (size_t)s * BB * GG, gh_s, h1, h1b,
        encout + (size_t)s * HH, (long long)SS * HH,
        encoutb + (size_t)s * HH, (long long)SS * HH);
  }
  // h0/h1 now hold encoder final states he0/he1 == decoder initial carry.

  // ---- precompute enc_out @ W_e^T + b_attn (W_e = W_attn[:, H:2H]) ----
  launch_gemm(encoutb, HH, wattnb + HH, 2 * HH, b_attn, encproj, HH,
              BB * SS, HH, HH, stream);

  // ---- decoder (teacher forcing) ----
  for (int t = 0; t < TT; ++t) {
    // hproj = h1 @ W_h^T (W_h = W_attn[:, 0:H]); no bias (bias folded in encproj)
    launch_gemm(h1b, HH, wattnb, 2 * HH, nullptr, hproj, HH, BB, HH, HH, stream);
    k_attn_scores<<<NB(BB * SS * 32), blk, 0, stream>>>(hproj, encproj, v_attn, scores);
    k_softmax_rows<<<1, 32, 0, stream>>>(scores, attw);
    k_attn_ctx<<<NB(BB * HH), blk, 0, stream>>>(attw, encout, ctx);
    k_build_rnn_in<<<NB(BB * (DD + HH)), blk, 0, stream>>>(
        decembb + (size_t)t * BB * DD, ctx, rnninb);

    // GRU d0
    launch_gemm(rnninb, DD + HH, wxd0b, DD + HH, bx_d0, gi_s, GG, BB, GG, DD + HH, stream);
    launch_gemm(h0b, HH, whd0b, HH, bh_d0, gh_s, GG, BB, GG, HH, stream);
    k_gru_gates<<<NB(BB * HH), blk, 0, stream>>>(gi_s, gh_s, h0, h0b,
                                                 nullptr, 0, nullptr, 0);
    // GRU d1 (input = new h0)
    launch_gemm(h0b, HH, wxd1b, HH, bx_d1, gi_s, GG, BB, GG, HH, stream);
    launch_gemm(h1b, HH, whd1b, HH, bh_d1, gh_s, GG, BB, GG, HH, stream);
    k_gru_gates<<<NB(BB * HH), blk, 0, stream>>>(gi_s, gh_s, h1, h1b,
                                                 nullptr, 0, nullptr, 0);

    // logits[b, t, :] = h1 @ W_out^T + b_out, written straight into d_out
    // C row stride = T*V so row b lands at out[b*T*V + t*V + n]
    launch_gemm(h1b, HH, woutb, HH, b_out, out + (size_t)t * VV,
                (long long)TT * VV, BB, VV, HH, stream);
  }
#undef NB
}
